// WindowAttention_78400333021343
// MI455X (gfx1250) — matrix operand order repaired and checked by reading, not compiled, not run on hardware
//
#include <hip/hip_runtime.h>
#include <hip/hip_bf16.h>

// ---------------------------------------------------------------------------
// Swin window attention, fused per-window, bf16 WMMA (gfx1250 / CDNA5).
//   N=4096 windows, L=49 (padded to 64), C=256, NH=8, dk=32, NW=64 win/img.
// ---------------------------------------------------------------------------

typedef __bf16 bf16;
typedef __attribute__((ext_vector_type(16))) __bf16 v16bf;
typedef __attribute__((ext_vector_type(8)))  __bf16 v8bf;
typedef __attribute__((ext_vector_type(4)))  __bf16 v4bf;
typedef __attribute__((ext_vector_type(8)))  float  v8f;

#define LPAD 64      // padded token count
#define CDIM 256
#define NHEAD 8
#define DK 32
#define LREAL 49
#define NWIN 64      // windows per image

static __device__ __forceinline__ bf16 f2bf(float f) {
    unsigned u = __builtin_bit_cast(unsigned, f);
    unsigned r = (u + 0x7FFFu + ((u >> 16) & 1u)) >> 16;   // RNE
    return __builtin_bit_cast(bf16, (unsigned short)r);
}
static __device__ __forceinline__ float bf2f(bf16 b) {
    unsigned u = ((unsigned)__builtin_bit_cast(unsigned short, b)) << 16;
    return __builtin_bit_cast(float, u);
}

// A-fragment (16-bit 16x32, MxK): lanes 0-15 -> K {kb..kb+7, kb+16..kb+23},
// lanes 16-31 -> K {kb+8..kb+15, kb+24..kb+31}.  Two 16B loads.
static __device__ __forceinline__ v16bf load_fragA(const bf16* base, int stride,
                                                   int row, int kbase, int lane) {
    const bf16* p = base + row * stride + kbase + ((lane & 16) ? 8 : 0);
    v8bf lo = *(const v8bf*)(p);
    v8bf hi = *(const v8bf*)(p + 16);
    v16bf r;
#pragma unroll
    for (int i = 0; i < 8; ++i) { r[i] = lo[i]; r[i + 8] = hi[i]; }
    return r;
}

// B-fragment (16-bit 32x16, KxN), source stored N-major so K is contiguous:
// lanes 0-15 -> K {kb..kb+7, kb+16..kb+23}, lanes 16-31 -> K {kb+8..kb+15, kb+24..kb+31}.  Two 16B loads.
static __device__ __forceinline__ v16bf load_fragB(const bf16* base, int stride,
                                                   int ncol, int kbase, int lane) {
    const bf16* p = base + ncol * stride + kbase + ((lane & 16) ? 8 : 0);
    v8bf lo = *(const v8bf*)(p);
    v8bf hi = *(const v8bf*)(p + 16);
    v16bf r;
#pragma unroll
    for (int i = 0; i < 8; ++i) { r[i] = lo[i]; r[i + 8] = hi[i]; }
    return r;
}

// ---------------------------------------------------------------------------
// Prep kernel A: wt[p][d][c] = bf16(W_p[c][d])  (transpose + downconvert)
// ---------------------------------------------------------------------------
__global__ void prep_weights(const float* __restrict__ wq, const float* __restrict__ wk,
                             const float* __restrict__ wv, const float* __restrict__ wo,
                             bf16* __restrict__ wt) {
    int idx = blockIdx.x * 256 + threadIdx.x;          // 4*256*256
    if (idx >= 4 * CDIM * CDIM) return;
    int p = idx >> 16, rem = idx & 65535;
    int d = rem >> 8, c = rem & 255;
    const float* w = (p == 0) ? wq : (p == 1) ? wk : (p == 2) ? wv : wo;
    wt[idx] = f2bf(w[c * CDIM + d]);
}

// ---------------------------------------------------------------------------
// Prep kernel B: addend[w][h][i][j] (64x8x64x64 f32), padded with -1e30.
// ---------------------------------------------------------------------------
__global__ void prep_addend(const float* __restrict__ mask,
                            const float* __restrict__ bias_table,
                            const int* __restrict__ rel_index,
                            float* __restrict__ addend) {
    int idx = blockIdx.x * 256 + threadIdx.x;          // 64*8*64*64
    if (idx >= NWIN * NHEAD * LPAD * LPAD) return;
    int w = idx >> 15, h = (idx >> 12) & 7, i = (idx >> 6) & 63, j = idx & 63;
    float v;
    if (j >= LREAL)      v = -1e30f;                   // padded key column
    else if (i >= LREAL) v = 0.0f;                     // padded query row (discarded)
    else v = bias_table[rel_index[i * LREAL + j] * NHEAD + h]
           + mask[(w * LREAL + i) * LREAL + j];
    addend[idx] = v;
}

// ---------------------------------------------------------------------------
// Main fused kernel: one workgroup (8 wave32) per window.
// LDS: s_x(32K, reused as s_o) | s_q(32K) | s_k(32K) | s_vt(32K) | s_p(64K)
// s_vt holds V transposed ([c][j]) so the P@V B-fragment is a contiguous load.
// ---------------------------------------------------------------------------
extern "C" __global__ void __launch_bounds__(256, 1)
win_attn_kernel(const float* __restrict__ x, const float* __restrict__ addend,
                const bf16* __restrict__ wt, float* __restrict__ out) {
    extern __shared__ __align__(16) char smem[];
    bf16* s_x  = (bf16*)smem;                // [64][256]  (phase5 reuses as s_o)
    bf16* s_q  = s_x  + LPAD * CDIM;         // [64][256]
    bf16* s_k  = s_q  + LPAD * CDIM;         // [64][256]
    bf16* s_vt = s_k  + LPAD * CDIM;         // [256][64]  V transposed
    bf16* s_p  = s_vt + CDIM * LPAD;         // [8][64][64]

    const int tid  = threadIdx.x;
    const int lane = tid & 31;
    const int wave = tid >> 5;
    const int win  = blockIdx.x;             // 0..4095
    const int wimg = win & (NWIN - 1);       // window index within image

    // ---- Phase 1: stage x window -> bf16 LDS (pad rows 49..63 with 0) ----
    {
        const float4* xg4 = (const float4*)(x + (size_t)win * LREAL * CDIM);
        for (int c4 = tid; c4 < (LPAD * CDIM) / 4; c4 += 256) {
            int row = c4 >> 6;                       // (c4*4)/256
            float4 v = (row < LREAL) ? xg4[c4] : make_float4(0.f, 0.f, 0.f, 0.f);
            v4bf d;
            d[0] = f2bf(v.x); d[1] = f2bf(v.y); d[2] = f2bf(v.z); d[3] = f2bf(v.w);
            *(v4bf*)(s_x + c4 * 4) = d;
        }
    }
    __syncthreads();

    // ---- Phase 2: Q/K/V = x @ W  (3 x [64x256], K=256); V stored transposed ----
    for (int t = wave; t < 3 * 64; t += 8) {
        int p = t >> 6, tile = t & 63;
        int mt = tile >> 4, nt = tile & 15;
        const bf16* wsel = wt + p * CDIM * CDIM;
        v8f acc = {};
#pragma unroll
        for (int kb = 0; kb < 8; ++kb) {
            v16bf a = load_fragA(s_x, CDIM, mt * 16 + (lane & 15), kb * 32, lane);
            v16bf b = load_fragB(wsel, CDIM, nt * 16 + (lane & 15), kb * 32, lane);
            acc = __builtin_amdgcn_wmma_f32_16x16x32_bf16(false, a, false, b,
                                                          (short)0, acc, false, false);
        }
        int rowbase = mt * 16 + ((lane & 16) ? 8 : 0);
        int col     = nt * 16 + (lane & 15);
        if (p == 2) {                                // V: transpose on store
#pragma unroll
            for (int r = 0; r < 8; ++r)
                s_vt[col * LPAD + rowbase + r] = f2bf(acc[r]);
        } else {
            bf16* dst = (p == 0) ? s_q : s_k;
#pragma unroll
            for (int r = 0; r < 8; ++r)
                dst[(rowbase + r) * CDIM + col] = f2bf(acc[r]);
        }
    }
    __syncthreads();

    // ---- Phase 3: scores S_h = Q_h K_h^T + addend, per head [64x64] ----
    for (int t = wave; t < NHEAD * 16; t += 8) {
        int h = t >> 4, tile = t & 15;
        int it = tile >> 2, jt = tile & 3;
        v16bf a = load_fragA(s_q, CDIM, it * 16 + (lane & 15), h * DK, lane);
        v16bf b = load_fragB(s_k, CDIM, jt * 16 + (lane & 15), h * DK, lane); // K^T
        v8f acc = {};
        acc = __builtin_amdgcn_wmma_f32_16x16x32_bf16(false, a, false, b,
                                                      (short)0, acc, false, false);
        const float* ad = addend + (((size_t)wimg * NHEAD + h) << 12);
        int rowbase = it * 16 + ((lane & 16) ? 8 : 0);
        int col     = jt * 16 + (lane & 15);
#pragma unroll
        for (int r = 0; r < 8; ++r) {
            float s = acc[r] + ad[(rowbase + r) * LPAD + col];
            s_p[(h << 12) + (rowbase + r) * LPAD + col] = f2bf(s);
        }
    }
    __syncthreads();

    // ---- Softmax over each of the 8*64 rows of s_p (vectorized 32B LDS ops) ----
    for (int rid = tid; rid < NHEAD * LPAD; rid += 256) {
        bf16* row = s_p + rid * LPAD;
        v16bf c[4];
#pragma unroll
        for (int q = 0; q < 4; ++q) c[q] = *(v16bf*)(row + q * 16);
        float f[LPAD];
        float mx = -3.4e38f;
#pragma unroll
        for (int q = 0; q < 4; ++q)
#pragma unroll
            for (int j = 0; j < 16; ++j) {
                float v = bf2f(c[q][j]);
                f[q * 16 + j] = v;
                mx = v > mx ? v : mx;
            }
        float sum = 0.0f;
#pragma unroll
        for (int j = 0; j < LPAD; ++j) { f[j] = __expf(f[j] - mx); sum += f[j]; }
        float inv = 1.0f / sum;
#pragma unroll
        for (int q = 0; q < 4; ++q) {
            v16bf o;
#pragma unroll
            for (int j = 0; j < 16; ++j) o[j] = f2bf(f[q * 16 + j] * inv);
            *(v16bf*)(row + q * 16) = o;
        }
    }
    __syncthreads();

    // ---- Phase 4: O_h = P_h @ V_h  ([64x64]@[64x32]) -> s_o (= s_x) ----
    bf16* s_o = s_x;
    for (int t = wave; t < NHEAD * 8; t += 8) {
        int h = t >> 3, tile = t & 7;
        int mt = tile >> 1, nt = tile & 1;
        int col = h * DK + nt * 16 + (lane & 15);
        v8f acc = {};
#pragma unroll
        for (int kb = 0; kb < 2; ++kb) {
            v16bf a = load_fragA(s_p + (h << 12), LPAD, mt * 16 + (lane & 15),
                                 kb * 32, lane);
            v16bf b = load_fragB(s_vt, LPAD, col, kb * 32, lane); // contiguous V^T
            acc = __builtin_amdgcn_wmma_f32_16x16x32_bf16(false, a, false, b,
                                                          (short)0, acc, false, false);
        }
        int rowbase = mt * 16 + ((lane & 16) ? 8 : 0);
#pragma unroll
        for (int r = 0; r < 8; ++r)
            s_o[(rowbase + r) * CDIM + col] = f2bf(acc[r]);
    }
    __syncthreads();

    // ---- Phase 5: out = O @ wo  ([64x256], K=256), store 49 real rows ----
    {
        const bf16* wo_t = wt + 3 * CDIM * CDIM;
        float* og = out + (size_t)win * LREAL * CDIM;
        for (int t = wave; t < 64; t += 8) {
            int mt = t >> 4, nt = t & 15;
            v8f acc = {};
#pragma unroll
            for (int kb = 0; kb < 8; ++kb) {
                v16bf a = load_fragA(s_o, CDIM, mt * 16 + (lane & 15), kb * 32, lane);
                v16bf b = load_fragB(wo_t, CDIM, nt * 16 + (lane & 15), kb * 32, lane);
                acc = __builtin_amdgcn_wmma_f32_16x16x32_bf16(false, a, false, b,
                                                              (short)0, acc, false, false);
            }
            int rowbase = mt * 16 + ((lane & 16) ? 8 : 0);
            int col     = nt * 16 + (lane & 15);
#pragma unroll
            for (int r = 0; r < 8; ++r) {
                int row = rowbase + r;
                if (row < LREAL) og[row * CDIM + col] = acc[r];
            }
        }
    }
}

// ---------------------------------------------------------------------------
// Launch
// ---------------------------------------------------------------------------
extern "C" void kernel_launch(void* const* d_in, const int* in_sizes, int n_in,
                              void* d_out, int out_size, void* d_ws, size_t ws_size,
                              hipStream_t stream) {
    const float* x          = (const float*)d_in[0];
    const float* mask       = (const float*)d_in[1];
    const float* wq         = (const float*)d_in[2];
    const float* wk         = (const float*)d_in[3];
    const float* wv         = (const float*)d_in[4];
    const float* wo         = (const float*)d_in[5];
    const float* bias_table = (const float*)d_in[6];
    const int*   rel_index  = (const int*)d_in[7];
    float* out = (float*)d_out;

    // Workspace: bf16 transposed weights (512 KB) + padded addend (8 MB)
    bf16*  wt     = (bf16*)d_ws;
    float* addend = (float*)((char*)d_ws + (size_t)4 * CDIM * CDIM * sizeof(bf16));

    prep_weights<<<(4 * CDIM * CDIM + 255) / 256, 256, 0, stream>>>(wq, wk, wv, wo, wt);
    prep_addend<<<(NWIN * NHEAD * LPAD * LPAD + 255) / 256, 256, 0, stream>>>(
        mask, bias_table, rel_index, addend);

    const size_t lds_bytes = (size_t)(4 * LPAD * CDIM + NHEAD * LPAD * LPAD) * sizeof(bf16);
    win_attn_kernel<<<4096, 256, lds_bytes, stream>>>(x, addend, wt, out);
}
